// HarmonicOscillator_10883447128671
// MI455X (gfx1250) — compile-verified, matrix operand order checked by reading, not run on hardware
//
#include <hip/hip_runtime.h>
#include <math.h>

// ---------------------------------------------------------------------------
// Harmonic oscillator additive synth for gfx1250 (MI455X).
//   B=8, C=1, T=251 frames, O=64 harmonics, L=48000 samples, out f32 [8,48000]
//
// Math restructure: phases[b,o,l] = (o+1) * Phi[b,l], Phi = cumsum(w_up).
//   -> only 8 scans of length 48000 (f64 accumulation, phase stored as
//      REVOLUTIONS Phi/2pi mod 1, so sin((o+1)Phi) = v_sin_f32((o+1)*rev))
//   -> harmonic reduction per 16-sample tile becomes S[16x64] x A[64x4frames]
//      matmul, executed as 2x V_WMMA_F32_16X16X32_F16 (f32 accum).
// ---------------------------------------------------------------------------

#define NB   8
#define NT   251
#define NO   64
#define NL   48000

typedef __attribute__((ext_vector_type(16))) _Float16 v16h;
typedef __attribute__((ext_vector_type(8)))  float    v8f;

__device__ __constant__ const float  POSC    = (float)(250.0 / 47999.0);       // (T-1)/(L-1)
__device__ __constant__ const float  W_SCALE = (float)(6.283185307179586 / 48000.0);

// ---------------------------------------------------------------------------
// Kernel A: masked, normalized, master-scaled amplitudes per (b, o, t).
// amps[b][o][t] = where(w*(o+1) > pi, 0, ov) / sum_o * master
// ---------------------------------------------------------------------------
__global__ __launch_bounds__(256) void amps_kernel(const float* __restrict__ f0,
                                                   const float* __restrict__ master,
                                                   const float* __restrict__ ov,
                                                   float* __restrict__ amps) {
  int idx = blockIdx.x * 256 + threadIdx.x;          // (b,t) pair
  if (idx >= NB * NT) return;
  int b = idx / NT, t = idx - b * NT;
  float w  = f0[idx] * W_SCALE;                       // rad/sample, f32 like ref
  float ma = master[idx];
  const float* src = ov   + ((size_t)b * NO) * NT + t;
  float*       dst = amps + ((size_t)b * NO) * NT + t;
  float av[NO];
  float sum = 0.0f;
#pragma unroll
  for (int o = 0; o < NO; ++o) {
    float a   = src[o * NT];
    float ofs = w * (float)(o + 1);
    a = (ofs > (float)M_PI) ? 0.0f : a;               // alias mask
    av[o] = a;
    sum  += a;
  }
  float scale = ma / sum;
#pragma unroll
  for (int o = 0; o < NO; ++o) dst[o * NT] = av[o] * scale;
}

// ---------------------------------------------------------------------------
// Kernel B: per-batch prefix sum of interpolated w; phase emitted as
// revolutions (Phi / 2pi mod 1) in f32.  One 1024-thread workgroup per batch:
// thread-local f64 partials + Hillis-Steele LDS scan, then a second pass.
// ---------------------------------------------------------------------------
__global__ __launch_bounds__(1024) void scan_kernel(const float* __restrict__ f0,
                                                    float* __restrict__ phir) {
  const int b   = blockIdx.x;
  const int tid = threadIdx.x;
  __shared__ float  wlds[NT];
  __shared__ double ssum[1024];

  for (int t = tid; t < NT; t += 1024) wlds[t] = f0[b * NT + t] * W_SCALE;
  __syncthreads();

  const int j0 = tid * 47;                            // 1024*47 = 48128 >= 48000
  const int j1 = (j0 + 47 < NL) ? (j0 + 47) : NL;
  double s = 0.0;
  for (int j = j0; j < j1; ++j) {
    float pos = (float)j * POSC;
    int   i0  = (int)pos; if (i0 > NT - 2) i0 = NT - 2;
    float fr  = pos - (float)i0;
    s += (double)(wlds[i0] * (1.0f - fr) + wlds[i0 + 1] * fr);
  }
  ssum[tid] = s;
  __syncthreads();
  for (int off = 1; off < 1024; off <<= 1) {          // inclusive scan
    double add = (tid >= off) ? ssum[tid - off] : 0.0;
    __syncthreads();
    ssum[tid] += add;
    __syncthreads();
  }
  double acc = (tid == 0) ? 0.0 : ssum[tid - 1];      // exclusive offset
  for (int j = j0; j < j1; ++j) {
    float pos = (float)j * POSC;
    int   i0  = (int)pos; if (i0 > NT - 2) i0 = NT - 2;
    float fr  = pos - (float)i0;
    acc += (double)(wlds[i0] * (1.0f - fr) + wlds[i0 + 1] * fr);
    double rev = acc * 0.15915494309189535;           // Phi / 2pi
    rev -= floor(rev);                                // mod 1 in f64
    phir[b * NL + j] = (float)rev;                    // (o+1)*rev <= 64
  }
}

// ---------------------------------------------------------------------------
// Kernel C: WMMA synthesis.  Each wave32 handles one 16-sample tile:
//   S[16 x 64] (sins via raw v_sin_f32 on revolutions, f16)
//   x A[64 x cols] (frame amps k0..k0+3, f16)  = D[16 x 16] f32,
//   done as two chained v_wmma_f32_16x16x32_f16.
// out[l] = (1-frac)*D[m, i0-k0] + frac*D[m, i0-k0+1]  (column pick via LDS).
// A-matrix per-lane layout (ISA 7.12.2, 16-bit A 16x32):
//   lane = 16*h + m (m=row, h=half); element e <-> K = e + 8h + 8*(e>=8).
// B-matrix mirrors with m = column index.
// ---------------------------------------------------------------------------
__global__ __launch_bounds__(256) void synth_kernel(const float* __restrict__ phir,
                                                    const float* __restrict__ amps,
                                                    float* __restrict__ out) {
  const int b    = blockIdx.y;
  const int tid  = threadIdx.x;
  const int wave = tid >> 5;
  const int lane = tid & 31;
  const int m    = lane & 15;        // row (sample in tile) for A; column for B
  const int h    = lane >> 4;        // K-half selector
  const int s0   = blockIdx.x * 128 + wave * 16;   // tile's first sample
  const int l    = s0 + m;                         // this lane's sample

  float rev = phir[b * NL + l];                    // phase in revolutions [0,1)

  // tile frame base + per-sample interp coords (f32, matches reference)
  float pos0 = (float)s0 * POSC;
  int   k0   = (int)pos0; if (k0 > NT - 2) k0 = NT - 2;
  float pos  = (float)l * POSC;
  int   i0   = (int)pos;  if (i0 > NT - 2) i0 = NT - 2;
  float frac = pos - (float)i0;

  const float* ampb = amps + ((size_t)b * NO) * NT;
  int kf = k0 + m; if (kf > NT - 1) kf = NT - 1;   // B column m's frame (cols>3 unused)

  v16h a0, a1, b0, b1;
#pragma unroll
  for (int e = 0; e < 16; ++e) {
    int K = e + 8 * h + ((e >> 3) << 3);           // 0..31
    // v_sin_f32(x) = sin(2*pi*x); hw range reduction covers |x| <= 64 here
    a0[e] = (_Float16)__builtin_amdgcn_sinf((float)(K + 1)  * rev);  // harm 1..32
    a1[e] = (_Float16)__builtin_amdgcn_sinf((float)(K + 33) * rev);  // harm 33..64
    b0[e] = (_Float16)ampb[K * NT + kf];
    b1[e] = (_Float16)ampb[(K + 32) * NT + kf];
  }

  v8f c = {};
  c = __builtin_amdgcn_wmma_f32_16x16x32_f16(false, a0, false, b0, (short)0, c, false, false);
  c = __builtin_amdgcn_wmma_f32_16x16x32_f16(false, a1, false, b1, (short)0, c, false, false);

  // spill D tile to LDS (row-padded, 33 dwords/row) for cross-lane column pick
  __shared__ float ldsD[8][264];                    // 8 waves x 8 rows x 33
#pragma unroll
  for (int r = 0; r < 8; ++r) ldsD[wave][r * 33 + lane] = c[r];  // D[r+8h][m]
  __syncthreads();

  if (h == 0) {                                     // 16 output lanes per wave
    int cidx = i0 - k0;                             // 0 or 1 within a 16-tile
    const float* row = &ldsD[wave][(m & 7) * 33 + (m >> 3) * 16];
    float D0 = row[cidx];
    float D1 = row[cidx + 1];
    out[b * NL + l] = (1.0f - frac) * D0 + frac * D1;
  }
}

// ---------------------------------------------------------------------------
extern "C" void kernel_launch(void* const* d_in, const int* in_sizes, int n_in,
                              void* d_out, int out_size, void* d_ws, size_t ws_size,
                              hipStream_t stream) {
  const float* f0     = (const float*)d_in[0];   // [8,1,251]
  const float* master = (const float*)d_in[1];   // [8,1,251]
  const float* ov     = (const float*)d_in[2];   // [8,1,64,251]
  float*       out    = (float*)d_out;           // [8,1,48000]

  float* phir = (float*)d_ws;                    // 8*48000 f32 (revolutions)
  float* amps = phir + NB * NL;                  // 8*64*251 f32  (~2.05 MB total)

  amps_kernel<<<(NB * NT + 255) / 256, 256, 0, stream>>>(f0, master, ov, amps);
  scan_kernel<<<NB, 1024, 0, stream>>>(f0, phir);
  dim3 g(NL / 128, NB, 1);                       // 375 x 8 blocks, 8 waves each
  synth_kernel<<<g, 256, 0, stream>>>(phir, amps, out);
}
